// ContactAttention_fix_em_5970004542129
// MI455X (gfx1250) — compile-verified, matrix operand order checked by reading, not run on hardware
//
#include <hip/hip_runtime.h>

#define L_SEQ 512
#define BATCH 2
#define DCH   127
#define EMB   128
#define NHEAD 2
#define HDIM  64
#define FFDIM 2048
#define EPSF  1e-5f

typedef __attribute__((ext_vector_type(16))) __bf16    v16bf;
typedef __attribute__((ext_vector_type(8)))  float     v8f;
typedef __attribute__((ext_vector_type(8)))  unsigned  v8u;

__device__ __forceinline__ unsigned short f2bf_bits(float f) {
  unsigned u = __builtin_bit_cast(unsigned, f);
  return (unsigned short)((u + 0x7FFFu + ((u >> 16) & 1u)) >> 16);  // RNE
}
__device__ __forceinline__ unsigned pack2(float lo, float hi) {
  return (unsigned)f2bf_bits(lo) | ((unsigned)f2bf_bits(hi) << 16);
}

// ---------------------------------------------------------------------------
// Generic batched GEMM: C = alpha * A(MxK) * B(KxN) [+bias] [relu]
// One wave (32 threads) computes one 16x16 tile via v_wmma_f32_16x16x32_bf16.
// A row-major (lda). TRANSB=1 -> Bm is NxK row-major; TRANSB=0 -> KxN.
// Batch: z = blockIdx.z, zo = z/zInner, zi = z%zInner select sub-matrices.
// M,N multiples of 16; K multiple of 32 (guaranteed by caller) -> no guards,
// EXEC stays all-ones for WMMA.
// Fragment layouts per ISA 7.12.2:
//   A: lane lr holds row M=lr; half=lane>>4 selects K {0-7,16-23}|{8-15,24-31},
//      VGPR v packs K pair (2 bf16) -> one float2 (b64) load per v.
//   C/D: lane lr = column N, VGPR r -> row r + 8*half.
// ---------------------------------------------------------------------------
template <int TRANSB>
__global__ __launch_bounds__(32)
void gemm_bf16_wmma(const float* __restrict__ A, const float* __restrict__ Bm,
                    const float* __restrict__ bias, float* __restrict__ C,
                    int K, int lda, int ldb, int ldc,
                    int relu, float alpha,
                    long sAo, long sAi, long sBo, long sBi, long sCo, long sCi,
                    int zInner) {
  int z  = blockIdx.z;
  int zo = z / zInner, zi = z % zInner;
  A  += zo * sAo + zi * sAi;
  Bm += zo * sBo + zi * sBi;
  C  += zo * sCo + zi * sCi;

  const int lane = threadIdx.x;
  const int half = lane >> 4;          // 0: K 0-7/16-23, 1: K 8-15/24-31
  const int lr   = lane & 15;
  const int tn = blockIdx.x * 16 + lr; // B/C column
  const int tm = blockIdx.y * 16 + lr; // A row

  // Hoisted per-lane base pointers (8-byte aligned: all offsets even floats).
  const float2* aRow = reinterpret_cast<const float2*>(A + (long)tm * lda + half * 8);
  const float2* bRow = nullptr;        // TRANSB path
  const float*  bCol = nullptr;        // non-TRANSB path
  if (TRANSB) bRow = reinterpret_cast<const float2*>(Bm + (long)tn * ldb + half * 8);
  else        bCol = Bm + tn;

  v8f acc = {};
  for (int k0 = 0; k0 < K; k0 += 32) {
    v8u au, bu;
    const int p0 = k0 >> 1;            // float2 index of this k-step
#pragma unroll
    for (int v = 0; v < 8; ++v) {
      const int pidx = p0 + (v & 3) + ((v >> 2) * 8);  // pair offsets {0..3, 8..11}
      float2 a2 = aRow[pidx];
      au[v] = pack2(a2.x, a2.y);
      if (TRANSB) {
        float2 b2 = bRow[pidx];
        bu[v] = pack2(b2.x, b2.y);
      } else {
        const int k = k0 + half * 8 + ((v < 4) ? 0 : 16) + 2 * (v & 3);
        bu[v] = pack2(bCol[(long)k * ldb], bCol[(long)(k + 1) * ldb]);
      }
    }
    v16bf a = __builtin_bit_cast(v16bf, au);
    v16bf b = __builtin_bit_cast(v16bf, bu);
    acc = __builtin_amdgcn_wmma_f32_16x16x32_bf16(false, a, false, b,
                                                  (short)0, acc, false, false);
  }

  float bv = bias ? bias[tn] : 0.f;
  float* cp = C + (long)(blockIdx.y * 16 + half * 8) * ldc + tn;
#pragma unroll
  for (int r = 0; r < 8; ++r) {
    float v = acc[r] * alpha + bv;
    if (relu) v = fmaxf(v, 0.f);
    cp[(long)r * ldc] = v;
  }
}

// ---------------------------------------------------------------------------
// Conv1d(4->127, k=9, dil=2, pad=8) + training BatchNorm + ReLU, fused.
// One block per output channel; block computes all B*L outputs + batch stats.
// seq: (B, L, 4) row-major.  Output s1: (B, 127, L).
// ---------------------------------------------------------------------------
__global__ __launch_bounds__(256)
void conv1_bn_relu(const float* __restrict__ seq, const float* __restrict__ w,
                   const float* __restrict__ cb, const float* __restrict__ g,
                   const float* __restrict__ beta, float* __restrict__ s1) {
  int d   = blockIdx.x;
  int tid = threadIdx.x;
  __shared__ float ws[36];
  __shared__ float red[256];
  __shared__ float stat[2];
  if (tid < 36) ws[tid] = w[d * 36 + tid];
  __syncthreads();

  float yv[4];
  float sum = 0.f, sq = 0.f;
#pragma unroll
  for (int seg = 0; seg < 4; ++seg) {
    int idx = tid + seg * 256;           // 0..1023 = (b,l)
    int b = idx >> 9, l = idx & 511;
    float y = cb[d];
    for (int t = 0; t < 9; ++t) {
      int lt = l + 2 * t - 8;
      if ((unsigned)lt < L_SEQ) {
        const float* sp = seq + ((long)b * L_SEQ + lt) * 4;
        y += ws[0 * 9 + t] * sp[0] + ws[1 * 9 + t] * sp[1] +
             ws[2 * 9 + t] * sp[2] + ws[3 * 9 + t] * sp[3];
      }
    }
    yv[seg] = y; sum += y; sq += y * y;
  }
  red[tid] = sum; __syncthreads();
  for (int rs = 128; rs > 0; rs >>= 1) { if (tid < rs) red[tid] += red[tid + rs]; __syncthreads(); }
  if (tid == 0) stat[0] = red[0] * (1.f / 1024.f);
  __syncthreads();
  red[tid] = sq; __syncthreads();
  for (int rs = 128; rs > 0; rs >>= 1) { if (tid < rs) red[tid] += red[tid + rs]; __syncthreads(); }
  if (tid == 0) stat[1] = red[0] * (1.f / 1024.f) - stat[0] * stat[0];
  __syncthreads();
  float mean = stat[0];
  float inv  = rsqrtf(stat[1] + EPSF);
  float gg = g[d], bb = beta[d];
#pragma unroll
  for (int seg = 0; seg < 4; ++seg) {
    int idx = tid + seg * 256;
    int b = idx >> 9, l = idx & 511;
    float v = gg * (yv[seg] - mean) * inv + bb;
    s1[((long)b * DCH + d) * L_SEQ + l] = fmaxf(v, 0.f);
  }
}

// ---------------------------------------------------------------------------
// Conv1d(127->127, k=9, dil=2, pad=8) + BN + ReLU. Writes transposed into the
// transformer activation x: (B, L, 128), channel d.
// ---------------------------------------------------------------------------
__global__ __launch_bounds__(256)
void conv2_bn_relu(const float* __restrict__ s1, const float* __restrict__ w,
                   const float* __restrict__ cb, const float* __restrict__ g,
                   const float* __restrict__ beta, float* __restrict__ x) {
  int d   = blockIdx.x;
  int tid = threadIdx.x;
  __shared__ float ws[DCH * 9];
  __shared__ float red[256];
  __shared__ float stat[2];
  for (int e = tid; e < DCH * 9; e += 256) ws[e] = w[(long)d * DCH * 9 + e];
  __syncthreads();

  float yv[4];
  float sum = 0.f, sq = 0.f;
#pragma unroll
  for (int seg = 0; seg < 4; ++seg) {
    int idx = tid + seg * 256;
    int b = idx >> 9, l = idx & 511;
    float y = cb[d];
    for (int t = 0; t < 9; ++t) {
      int lt = l + 2 * t - 8;
      if ((unsigned)lt >= L_SEQ) continue;
      const float* sp = s1 + (long)b * DCH * L_SEQ + lt;
      const float* wp = ws + t;
      float acc = 0.f;
      for (int c = 0; c < DCH; ++c)
        acc += wp[c * 9] * sp[(long)c * L_SEQ];
      y += acc;
    }
    yv[seg] = y; sum += y; sq += y * y;
  }
  red[tid] = sum; __syncthreads();
  for (int rs = 128; rs > 0; rs >>= 1) { if (tid < rs) red[tid] += red[tid + rs]; __syncthreads(); }
  if (tid == 0) stat[0] = red[0] * (1.f / 1024.f);
  __syncthreads();
  red[tid] = sq; __syncthreads();
  for (int rs = 128; rs > 0; rs >>= 1) { if (tid < rs) red[tid] += red[tid + rs]; __syncthreads(); }
  if (tid == 0) stat[1] = red[0] * (1.f / 1024.f) - stat[0] * stat[0];
  __syncthreads();
  float mean = stat[0];
  float inv  = rsqrtf(stat[1] + EPSF);
  float gg = g[d], bb = beta[d];
#pragma unroll
  for (int seg = 0; seg < 4; ++seg) {
    int idx = tid + seg * 256;
    int b = idx >> 9, l = idx & 511;
    float v = gg * (yv[seg] - mean) * inv + bb;
    x[((long)b * L_SEQ + l) * EMB + d] = fmaxf(v, 0.f);
  }
}

__global__ void set_pos_channel(float* __restrict__ x) {
  int idx = blockIdx.x * blockDim.x + threadIdx.x;  // (b,l)
  if (idx >= BATCH * L_SEQ) return;
  int l = idx & (L_SEQ - 1);
  x[(long)idx * EMB + (EMB - 1)] = (float)(l + 1) * (1.0f / L_SEQ);
}

// x = LayerNorm(x + o) over last dim (128), in place. One block per row.
__global__ __launch_bounds__(EMB)
void ln_add_inplace(float* __restrict__ x, const float* __restrict__ o,
                    const float* __restrict__ g, const float* __restrict__ beta) {
  int r = blockIdx.x, t = threadIdx.x;
  float y = x[(long)r * EMB + t] + o[(long)r * EMB + t];
  __shared__ float red[EMB];
  __shared__ float stat[2];
  red[t] = y; __syncthreads();
  for (int rs = EMB / 2; rs > 0; rs >>= 1) { if (t < rs) red[t] += red[t + rs]; __syncthreads(); }
  if (t == 0) stat[0] = red[0] * (1.0f / EMB);
  __syncthreads();
  float c = y - stat[0];
  red[t] = c * c; __syncthreads();
  for (int rs = EMB / 2; rs > 0; rs >>= 1) { if (t < rs) red[t] += red[t + rs]; __syncthreads(); }
  if (t == 0) stat[1] = red[0] * (1.0f / EMB);
  __syncthreads();
  x[(long)r * EMB + t] = g[t] * c * rsqrtf(stat[1] + EPSF) + beta[t];
}

// Row softmax over 512 columns, in place. One block (256 thr) per row.
__global__ __launch_bounds__(256)
void softmax_rows(float* __restrict__ s) {
  int row = blockIdx.x;
  int t   = threadIdx.x;
  float* p = s + (long)row * L_SEQ;
  float v0 = p[t], v1 = p[t + 256];
  __shared__ float red[256];
  __shared__ float stat;
  red[t] = fmaxf(v0, v1); __syncthreads();
  for (int q = 128; q > 0; q >>= 1) { if (t < q) red[t] = fmaxf(red[t], red[t + q]); __syncthreads(); }
  if (t == 0) stat = red[0];
  __syncthreads();
  float m = stat;
  float e0 = __expf(v0 - m), e1 = __expf(v1 - m);
  red[t] = e0 + e1; __syncthreads();
  for (int q = 128; q > 0; q >>= 1) { if (t < q) red[t] += red[t + q]; __syncthreads(); }
  if (t == 0) stat = red[0];
  __syncthreads();
  float inv = 1.f / stat;
  p[t] = e0 * inv; p[t + 256] = e1 * inv;
}

// ---------------------------------------------------------------------------
// Fused pairwise representation + LocallyConnected2d(k=1):
// pre[b,i,j] = sum_c emb[b,c,min(i,j)]*lcw[c,i,j]
//            + sum_c emb[b,c,max(i,j)]*lcw[128+c,i,j]
//            + r[i]*lcw[256,i,j] + r[j]*lcw[257,i,j],  r[k]=(k+1)/L
// emb[b,c,l] == x[b,l,c]. 32x32 pixel tile per block; emb columns in LDS;
// lcw reads coalesced along j (the 540 MB stream that dominates runtime).
// ---------------------------------------------------------------------------
__global__ __launch_bounds__(256)
void contact_lc(const float* __restrict__ x, const float* __restrict__ lcw,
                float* __restrict__ pre) {
  int b  = blockIdx.z;
  int j0 = blockIdx.x * 32, i0 = blockIdx.y * 32;
  int tx = threadIdx.x, ty = threadIdx.y;
  int tid = ty * 32 + tx;
  __shared__ float embI[EMB * 32];
  __shared__ float embJ[EMB * 32];
  for (int e = tid; e < EMB * 32; e += 256) {
    int c = e >> 5, idx = e & 31;
    embJ[e] = x[((long)b * L_SEQ + j0 + idx) * EMB + c];
    embI[e] = x[((long)b * L_SEQ + i0 + idx) * EMB + c];
  }
  __syncthreads();
  const long LL = (long)L_SEQ * L_SEQ;
#pragma unroll
  for (int s = 0; s < 4; ++s) {
    int ii = ty + 8 * s;
    int i  = i0 + ii;
    int j  = j0 + tx;
    bool lower = (j <= i);
    const float* lw = lcw + (long)i * L_SEQ + j;
    float acc = 0.f;
    for (int c = 0; c < EMB; ++c) {
      float vj = embJ[c * 32 + tx];
      float vi = embI[c * 32 + ii];
      float eJ = lower ? vj : vi;   // emb at min(i,j)
      float eI = lower ? vi : vj;   // emb at max(i,j)
      acc += eJ * lw[(long)c * LL] + eI * lw[(long)(EMB + c) * LL];
    }
    acc += (float)(i + 1) * (1.0f / L_SEQ) * lw[(long)(2 * EMB) * LL];
    acc += (float)(j + 1) * (1.0f / L_SEQ) * lw[(long)(2 * EMB + 1) * LL];
    pre[((long)b * L_SEQ + i) * L_SEQ + j] = acc;
  }
}

__global__ void symmetrize(const float* __restrict__ pre, float* __restrict__ out) {
  long idx = (long)blockIdx.x * blockDim.x + threadIdx.x;
  const long LL = (long)L_SEQ * L_SEQ;
  if (idx >= (long)BATCH * LL) return;
  long b = idx / LL, rem = idx % LL;
  long i = rem / L_SEQ, j = rem % L_SEQ;
  out[idx] = 0.5f * (pre[idx] + pre[b * LL + j * L_SEQ + i]);
}

// ---------------------------------------------------------------------------
extern "C" void kernel_launch(void* const* d_in, const int* in_sizes, int n_in,
                              void* d_out, int out_size, void* d_ws, size_t ws_size,
                              hipStream_t stream) {
  (void)in_sizes; (void)n_in; (void)out_size; (void)ws_size;
  const float* seq       = (const float*)d_in[1];
  const float* conv1_w   = (const float*)d_in[3];
  const float* conv1_b   = (const float*)d_in[4];
  const float* bn1_g     = (const float*)d_in[5];
  const float* bn1_b     = (const float*)d_in[6];
  const float* conv2_w   = (const float*)d_in[7];
  const float* conv2_b   = (const float*)d_in[8];
  const float* bn2_g     = (const float*)d_in[9];
  const float* bn2_b     = (const float*)d_in[10];
  const float* attn_in_w = (const float*)d_in[11];
  const float* attn_in_b = (const float*)d_in[12];
  const float* attn_out_w= (const float*)d_in[13];
  const float* attn_out_b= (const float*)d_in[14];
  const float* ln1_g     = (const float*)d_in[15];
  const float* ln1_b     = (const float*)d_in[16];
  const float* ffn_w1    = (const float*)d_in[17];
  const float* ffn_b1    = (const float*)d_in[18];
  const float* ffn_w2    = (const float*)d_in[19];
  const float* ffn_b2    = (const float*)d_in[20];
  const float* ln2_g     = (const float*)d_in[21];
  const float* ln2_b     = (const float*)d_in[22];
  const float* lc_w      = (const float*)d_in[23];
  float* out = (float*)d_out;

  float* ws  = (float*)d_ws;
  float* s1  = ws;  ws += (long)BATCH * DCH * L_SEQ;            // 130048
  float* x   = ws;  ws += (long)BATCH * L_SEQ * EMB;            // 131072
  float* qkv = ws;  ws += (long)BATCH * L_SEQ * 3 * EMB;        // 393216
  float* sc  = ws;  ws += (long)BATCH * NHEAD * L_SEQ * L_SEQ;  // 1048576
  float* oo  = ws;  ws += (long)BATCH * L_SEQ * EMB;
  float* tmp = ws;  ws += (long)BATCH * L_SEQ * EMB;
  float* ffh = ws;  ws += (long)BATCH * L_SEQ * FFDIM;          // 2097152
  float* pre = ws;  ws += (long)BATCH * L_SEQ * L_SEQ;          // 524288

  conv1_bn_relu<<<DCH, 256, 0, stream>>>(seq, conv1_w, conv1_b, bn1_g, bn1_b, s1);
  conv2_bn_relu<<<DCH, 256, 0, stream>>>(s1, conv2_w, conv2_b, bn2_g, bn2_b, x);
  set_pos_channel<<<(BATCH * L_SEQ + 255) / 256, 256, 0, stream>>>(x);

  const int M = BATCH * L_SEQ;  // 1024 rows, x treated as (M, 128)
  for (int i = 0; i < 3; ++i) {
    const float* iw = attn_in_w  + (long)i * 3 * EMB * EMB;
    const float* ib = attn_in_b  + (long)i * 3 * EMB;
    const float* ow = attn_out_w + (long)i * EMB * EMB;
    const float* ob = attn_out_b + (long)i * EMB;

    // qkv = x @ iw^T + ib : (1024 x 384)
    gemm_bf16_wmma<1><<<dim3(3 * EMB / 16, M / 16, 1), 32, 0, stream>>>(
        x, iw, ib, qkv, EMB, EMB, EMB, 3 * EMB,
        /*relu*/0, 1.0f, 0, 0, 0, 0, 0, 0, 1);

    // scores[b,h] = Q K^T / sqrt(64) : per (b,h) 512x512, K=64
    gemm_bf16_wmma<1><<<dim3(L_SEQ / 16, L_SEQ / 16, BATCH * NHEAD), 32, 0, stream>>>(
        qkv, qkv + EMB, nullptr, sc, HDIM, 3 * EMB, 3 * EMB, L_SEQ,
        0, 0.125f,
        (long)L_SEQ * 3 * EMB, HDIM,
        (long)L_SEQ * 3 * EMB, HDIM,
        (long)NHEAD * L_SEQ * L_SEQ, (long)L_SEQ * L_SEQ, NHEAD);

    softmax_rows<<<BATCH * NHEAD * L_SEQ, 256, 0, stream>>>(sc);

    // o[b,h] = P @ V : 512x64, K=512; heads concatenated into (B,L,128)
    gemm_bf16_wmma<0><<<dim3(HDIM / 16, L_SEQ / 16, BATCH * NHEAD), 32, 0, stream>>>(
        sc, qkv + 2 * EMB, nullptr, oo, L_SEQ, L_SEQ, 3 * EMB, EMB,
        0, 1.0f,
        (long)NHEAD * L_SEQ * L_SEQ, (long)L_SEQ * L_SEQ,
        (long)L_SEQ * 3 * EMB, HDIM,
        (long)L_SEQ * EMB, HDIM, NHEAD);

    // out-proj
    gemm_bf16_wmma<1><<<dim3(EMB / 16, M / 16, 1), 32, 0, stream>>>(
        oo, ow, ob, tmp, EMB, EMB, EMB, EMB,
        0, 1.0f, 0, 0, 0, 0, 0, 0, 1);
    ln_add_inplace<<<M, EMB, 0, stream>>>(x, tmp, ln1_g + i * EMB, ln1_b + i * EMB);

    // FFN
    gemm_bf16_wmma<1><<<dim3(FFDIM / 16, M / 16, 1), 32, 0, stream>>>(
        x, ffn_w1 + (long)i * FFDIM * EMB, ffn_b1 + i * FFDIM, ffh,
        EMB, EMB, EMB, FFDIM, /*relu*/1, 1.0f, 0, 0, 0, 0, 0, 0, 1);
    gemm_bf16_wmma<1><<<dim3(EMB / 16, M / 16, 1), 32, 0, stream>>>(
        ffh, ffn_w2 + (long)i * EMB * FFDIM, ffn_b2 + i * EMB, tmp,
        FFDIM, FFDIM, FFDIM, EMB, 0, 1.0f, 0, 0, 0, 0, 0, 0, 1);
    ln_add_inplace<<<M, EMB, 0, stream>>>(x, tmp, ln2_g + i * EMB, ln2_b + i * EMB);
  }

  contact_lc<<<dim3(L_SEQ / 32, L_SEQ / 32, BATCH), dim3(32, 8), 0, stream>>>(x, lc_w, pre);
  symmetrize<<<(BATCH * L_SEQ * L_SEQ + 255) / 256, 256, 0, stream>>>(pre, out);
}
